// minLSTMCell_523986010155
// MI455X (gfx1250) — matrix operand order repaired and checked by reading, not compiled, not run on hardware
//
#include <hip/hip_runtime.h>
#include <math.h>

// Problem constants (match reference): B=4, S=4096, D=1024, H=1024
#define BB 4
#define SS 4096
#define DD 1024
#define HH 1024

typedef __attribute__((ext_vector_type(16))) __bf16 v16bf;
typedef __attribute__((ext_vector_type(8)))  __bf16 v8bf;
typedef __attribute__((ext_vector_type(8)))  float  v8f;

// D = A*B + C, bf16 inputs, f32 accumulate (CDNA5 V_WMMA_F32_16X16X32_BF16)
#define WMMA_BF16(A, Bm, C) \
  __builtin_amdgcn_wmma_f32_16x16x32_bf16(false, (A), false, (Bm), (short)0, (C), false, false)

__device__ __forceinline__ float softplus_f(float x) {
  return fmaxf(x, 0.0f) + log1pf(__expf(-fabsf(x)));
}
__device__ __forceinline__ float log_g_f(float x) {
  return (x >= 0.0f) ? __logf(x + 0.5f) : -softplus_f(-x);
}

// ---------------------------------------------------------------------------
// Kernel 0: one-time f32 -> bf16 hi/lo split (bf16x2 GEMM emulation operands).
// Moves all conversion VALU out of the GEMM inner loop.
// ---------------------------------------------------------------------------
__global__ __launch_bounds__(256)
void split_bf16_kernel(const float* __restrict__ in, __bf16* __restrict__ hi,
                       __bf16* __restrict__ lo, int n8) {
  int i = blockIdx.x * 256 + threadIdx.x;
  if (i >= n8) return;
  const float4* p = reinterpret_cast<const float4*>(in) + (size_t)i * 2;
  float4 t0 = p[0], t1 = p[1];
  float v[8] = {t0.x, t0.y, t0.z, t0.w, t1.x, t1.y, t1.z, t1.w};
  v8bf h, l;
#pragma unroll
  for (int e = 0; e < 8; ++e) {
    __bf16 hh = (__bf16)v[e];
    h[e] = hh;
    l[e] = (__bf16)(v[e] - (float)hh);
  }
  *reinterpret_cast<v8bf*>(hi + (size_t)i * 8) = h;
  *reinterpret_cast<v8bf*>(lo + (size_t)i * 8) = l;
}

// A operand (16x32 bf16): per-lane K = half*8+{0..7}, 16+half*8+{0..7} -> two 16B loads
__device__ __forceinline__ v16bf load_A(const __bf16* __restrict__ row, int kb, int half) {
  v8bf c0 = *reinterpret_cast<const v8bf*>(row + kb + half * 8);
  v8bf c1 = *reinterpret_cast<const v8bf*>(row + kb + 16 + half * 8);
  return __builtin_shufflevector(c0, c1, 0, 1, 2, 3, 4, 5, 6, 7, 8, 9, 10, 11, 12, 13, 14, 15);
}
// B operand (32x16 bf16): per-lane column N = lane%16, K = half*8+{0..7}, 16+half*8+{0..7} -> two 16B loads
__device__ __forceinline__ v16bf load_B(const __bf16* __restrict__ row, int kb, int half) {
  v8bf c0 = *reinterpret_cast<const v8bf*>(row + kb + half * 8);
  v8bf c1 = *reinterpret_cast<const v8bf*>(row + kb + 16 + half * 8);
  return __builtin_shufflevector(c0, c1, 0, 1, 2, 3, 4, 5, 6, 7, 8, 9, 10, 11, 12, 13, 14, 15);
}

// Fused gate epilogue for one 16x16 tile (8 values per lane) -> log_f, log_v in (B,H,S) layout
__device__ __forceinline__ void epilogue_store(const v8f f, const v8f i, const v8f h,
                                               float bF, float bI, float bH,
                                               float* __restrict__ lf_p,
                                               float* __restrict__ lv_p) {
  float of[8], ov[8];
#pragma unroll
  for (int r = 0; r < 8; ++r) {
    float fp = f[r] + bF;
    float ip = i[r] + bI;
    float hp = h[r] + bH;
    float diff = softplus_f(-fp) - softplus_f(-ip);
    of[r] = -softplus_f(diff);                 // log_f
    ov[r] = -softplus_f(-diff) + log_g_f(hp);  // log_i + log_tilde_h
  }
  float4* lf4 = reinterpret_cast<float4*>(lf_p);
  float4* lv4 = reinterpret_cast<float4*>(lv_p);
  lf4[0] = make_float4(of[0], of[1], of[2], of[3]);
  lf4[1] = make_float4(of[4], of[5], of[6], of[7]);
  lv4[0] = make_float4(ov[0], ov[1], ov[2], ov[3]);
  lv4[1] = make_float4(ov[4], ov[5], ov[6], ov[7]);
}

// ---------------------------------------------------------------------------
// Kernel 1: triple GEMM (x*W_f^T, x*W_i^T, x*W_h^T) via bf16x2 WMMA + fused
// gate math. Wave tile: 64(M) x 16(N), 3 gates => 36 WMMAs per K=32 step.
// Block = 4 waves covering 64x64.
// ---------------------------------------------------------------------------
__global__ __launch_bounds__(128)
void gates_wmma_kernel(const __bf16* __restrict__ xh, const __bf16* __restrict__ xl,
                       const __bf16* __restrict__ Wfh, const __bf16* __restrict__ Wfl,
                       const __bf16* __restrict__ Wih, const __bf16* __restrict__ Wil,
                       const __bf16* __restrict__ Whh, const __bf16* __restrict__ Whl,
                       const float* __restrict__ bf, const float* __restrict__ bi,
                       const float* __restrict__ bh,
                       float* __restrict__ lf_out, float* __restrict__ lv_out) {
  const int lane = threadIdx.x & 31;
  const int wave = threadIdx.x >> 5;
  const int l16  = lane & 15;
  const int half = lane >> 4;
  const int n    = blockIdx.x * 64 + wave * 16 + l16;  // per-lane output column (h index)
  const int m0   = blockIdx.y * 64;                    // row tile base in B*S

  const __bf16* a0h = xh + (size_t)(m0 + l16) * DD;
  const __bf16* a0l = xl + (size_t)(m0 + l16) * DD;
  const size_t rstep = (size_t)16 * DD;
  const __bf16* wfhr = Wfh + (size_t)n * DD;
  const __bf16* wflr = Wfl + (size_t)n * DD;
  const __bf16* wihr = Wih + (size_t)n * DD;
  const __bf16* wilr = Wil + (size_t)n * DD;
  const __bf16* whhr = Whh + (size_t)n * DD;
  const __bf16* whlr = Whl + (size_t)n * DD;

  v8f z = {0.f, 0.f, 0.f, 0.f, 0.f, 0.f, 0.f, 0.f};
  v8f cF0 = z, cF1 = z, cF2 = z, cF3 = z;
  v8f cI0 = z, cI1 = z, cI2 = z, cI3 = z;
  v8f cH0 = z, cH1 = z, cH2 = z, cH3 = z;

  for (int kb = 0; kb < DD; kb += 32) {
    // A operands: 4 row-tiles, hi+lo
    v16bf A0h = load_A(a0h, kb, half),             A0l = load_A(a0l, kb, half);
    v16bf A1h = load_A(a0h + rstep, kb, half),     A1l = load_A(a0l + rstep, kb, half);
    v16bf A2h = load_A(a0h + 2 * rstep, kb, half), A2l = load_A(a0l + 2 * rstep, kb, half);
    v16bf A3h = load_A(a0h + 3 * rstep, kb, half), A3l = load_A(a0l + 3 * rstep, kb, half);

    // Gate F
    {
      v16bf Bh = load_B(wfhr, kb, half), Bl = load_B(wflr, kb, half);
      cF0 = WMMA_BF16(A0h, Bh, cF0); cF0 = WMMA_BF16(A0h, Bl, cF0); cF0 = WMMA_BF16(A0l, Bh, cF0);
      cF1 = WMMA_BF16(A1h, Bh, cF1); cF1 = WMMA_BF16(A1h, Bl, cF1); cF1 = WMMA_BF16(A1l, Bh, cF1);
      cF2 = WMMA_BF16(A2h, Bh, cF2); cF2 = WMMA_BF16(A2h, Bl, cF2); cF2 = WMMA_BF16(A2l, Bh, cF2);
      cF3 = WMMA_BF16(A3h, Bh, cF3); cF3 = WMMA_BF16(A3h, Bl, cF3); cF3 = WMMA_BF16(A3l, Bh, cF3);
    }
    // Gate I
    {
      v16bf Bh = load_B(wihr, kb, half), Bl = load_B(wilr, kb, half);
      cI0 = WMMA_BF16(A0h, Bh, cI0); cI0 = WMMA_BF16(A0h, Bl, cI0); cI0 = WMMA_BF16(A0l, Bh, cI0);
      cI1 = WMMA_BF16(A1h, Bh, cI1); cI1 = WMMA_BF16(A1h, Bl, cI1); cI1 = WMMA_BF16(A1l, Bh, cI1);
      cI2 = WMMA_BF16(A2h, Bh, cI2); cI2 = WMMA_BF16(A2h, Bl, cI2); cI2 = WMMA_BF16(A2l, Bh, cI2);
      cI3 = WMMA_BF16(A3h, Bh, cI3); cI3 = WMMA_BF16(A3h, Bl, cI3); cI3 = WMMA_BF16(A3l, Bh, cI3);
    }
    // Gate H
    {
      v16bf Bh = load_B(whhr, kb, half), Bl = load_B(whlr, kb, half);
      cH0 = WMMA_BF16(A0h, Bh, cH0); cH0 = WMMA_BF16(A0h, Bl, cH0); cH0 = WMMA_BF16(A0l, Bh, cH0);
      cH1 = WMMA_BF16(A1h, Bh, cH1); cH1 = WMMA_BF16(A1h, Bl, cH1); cH1 = WMMA_BF16(A1l, Bh, cH1);
      cH2 = WMMA_BF16(A2h, Bh, cH2); cH2 = WMMA_BF16(A2h, Bl, cH2); cH2 = WMMA_BF16(A2l, Bh, cH2);
      cH3 = WMMA_BF16(A3h, Bh, cH3); cH3 = WMMA_BF16(A3h, Bl, cH3); cH3 = WMMA_BF16(A3l, Bh, cH3);
    }
  }

  const float vbF = bf[n], vbI = bi[n], vbH = bh[n];
  const int b     = m0 / SS;  // 64 | 4096 -> uniform per tile
  const int sBase = m0 - b * SS;
  // (B,H,S) layout: each lane's 8 acc values are 8 consecutive s -> 2x b128 stores
  float* lf_p = lf_out + ((size_t)b * HH + n) * SS + sBase + half * 8;
  float* lv_p = lv_out + ((size_t)b * HH + n) * SS + sBase + half * 8;
  epilogue_store(cF0, cI0, cH0, vbF, vbI, vbH, lf_p,      lv_p);
  epilogue_store(cF1, cI1, cH1, vbF, vbI, vbH, lf_p + 16, lv_p + 16);
  epilogue_store(cF2, cI2, cH2, vbF, vbI, vbH, lf_p + 32, lv_p + 32);
  epilogue_store(cF3, cI3, cH3, vbF, vbI, vbH, lf_p + 48, lv_p + 48);
}

// ---------------------------------------------------------------------------
// Kernel 2: per-(b,h) channel log-space scan. One wave per channel, LDS-staged,
// three-phase hierarchical scan (lane chunks of 128, __shfl_up wave scans).
// ---------------------------------------------------------------------------
#define SCAN_WAVES 2
__global__ __launch_bounds__(64)
void scan_kernel(const float* __restrict__ lf, const float* __restrict__ lv,
                 const float* __restrict__ pre_h, float* __restrict__ out) {
  __shared__ float sF[SCAN_WAVES][SS];
  __shared__ float sV[SCAN_WAVES][SS];

  const int lane = threadIdx.x & 31;
  const int wave = threadIdx.x >> 5;
  const int ch   = blockIdx.x * SCAN_WAVES + wave;  // = b*H + h
  const int b    = ch / HH;
  const int h    = ch - b * HH;

  const float4* F4 = reinterpret_cast<const float4*>(lf + (size_t)ch * SS);
  const float4* V4 = reinterpret_cast<const float4*>(lv + (size_t)ch * SS);
  float4* sF4 = reinterpret_cast<float4*>(&sF[wave][0]);
  float4* sV4 = reinterpret_cast<float4*>(&sV[wave][0]);
#pragma unroll 4
  for (int j = lane; j < SS / 4; j += 32) { sF4[j] = F4[j]; sV4[j] = V4[j]; }
  __syncthreads();

  // Sequence t = 0..S: astar[t] = sum_{u<=t} F[u] for t<S, astar[S] = 0 (reference end-pad);
  // lv[0]=log_g(pre_h), lv[t]=V[t-1]; out[t-1]=exp(astar[t]+cumlogsumexp_t).
  const int tb = lane * 128 + 1;

  // Pass 1: lane sums of F over u = tb..tb+127 (u==S excluded), wave scan -> cumsum carry
  float q = 0.f;
#pragma unroll 4
  for (int j = 0; j < 128; ++j) {
    int u = tb + j;
    if (u < SS) q += sF[wave][u];
  }
  float inc = q;
  for (int d = 1; d < 32; d <<= 1) {
    float t = __shfl_up(inc, d, 32);
    if (lane >= d) inc += t;
  }
  const float baseA = sF[wave][0] + (inc - q);  // sum_{u=0}^{lane*128} F[u]

  // Pass 2: lane-local logsumexp of z = lv[t] - astar[t]
  float mx = -INFINITY, sm = 0.f;
  {
    float a = baseA;
#pragma unroll 4
    for (int j = 0; j < 128; ++j) {
      int t = tb + j;
      if (t < SS) a += sF[wave][t];
      float as = (t == SS) ? 0.f : a;
      float zz = sV[wave][t - 1] - as;
      if (zz <= mx) { sm += __expf(zz - mx); }
      else          { sm = sm * __expf(mx - zz) + 1.f; mx = zz; }
    }
  }
  // Wave inclusive scan of (max,sum) pairs, shift to exclusive
  float m = mx, s = sm;
  for (int d = 1; d < 32; d <<= 1) {
    float m2 = __shfl_up(m, d, 32);
    float s2 = __shfl_up(s, d, 32);
    if (lane >= d) {
      float nm = fmaxf(m, m2);
      s = s * __expf(m - nm) + s2 * __expf(m2 - nm);
      m = nm;
    }
  }
  float me = __shfl_up(m, 1, 32);
  float se = __shfl_up(s, 1, 32);
  if (lane == 0) { me = -INFINITY; se = 0.f; }
  // Fold the t=0 element: z0 = log_g(pre_h) - astar[0] (astar[0] = F[0])
  const float z0 = log_g_f(pre_h[ch]) - sF[wave][0];
  float mc = fmaxf(z0, me);
  float sc = __expf(z0 - mc) + se * __expf(me - mc);

  // Pass 3: stream final outputs: out[b, t-1, h] = exp(astar[t] + m[t])
  {
    float a = baseA;
    float* outp = out + (size_t)b * SS * HH + h;
#pragma unroll 2
    for (int j = 0; j < 128; ++j) {
      int t = tb + j;
      if (t < SS) a += sF[wave][t];
      float as = (t == SS) ? 0.f : a;
      float zz = sV[wave][t - 1] - as;
      if (zz <= mc) { sc += __expf(zz - mc); }
      else          { sc = sc * __expf(mc - zz) + 1.f; mc = zz; }
      outp[(size_t)(t - 1) * HH] = __expf(as + mc + __logf(sc));
    }
  }
}

extern "C" void kernel_launch(void* const* d_in, const int* in_sizes, int n_in,
                              void* d_out, int out_size, void* d_ws, size_t ws_size,
                              hipStream_t stream) {
  (void)in_sizes; (void)n_in; (void)out_size; (void)ws_size;
  const float* x    = (const float*)d_in[0];
  const float* preh = (const float*)d_in[1];
  const float* Wf   = (const float*)d_in[2];
  const float* bf   = (const float*)d_in[3];
  const float* Wi   = (const float*)d_in[4];
  const float* bi   = (const float*)d_in[5];
  const float* Wh   = (const float*)d_in[6];
  const float* bh   = (const float*)d_in[7];
  float* out = (float*)d_out;

  const size_t NX = (size_t)BB * SS * HH;  // 16M elements (also = B*S*D)
  const size_t NW = (size_t)HH * DD;       // 1M elements

  // Workspace layout: lf, lv (f32, 64MB each), then bf16 hi/lo splits (76MB)
  float*  lf  = (float*)d_ws;
  float*  lv  = lf + NX;
  __bf16* xhi = (__bf16*)(lv + NX);
  __bf16* xlo = xhi + NX;
  __bf16* wfh = xlo + NX;
  __bf16* wfl = wfh + NW;
  __bf16* wih = wfl + NW;
  __bf16* wil = wih + NW;
  __bf16* whh = wil + NW;
  __bf16* whl = whh + NW;

  // Phase 0: one-time bf16x2 operand splits
  split_bf16_kernel<<<dim3((unsigned)(NX / 8 / 256)), dim3(256), 0, stream>>>(x, xhi, xlo, (int)(NX / 8));
  split_bf16_kernel<<<dim3((unsigned)(NW / 8 / 256)), dim3(256), 0, stream>>>(Wf, wfh, wfl, (int)(NW / 8));
  split_bf16_kernel<<<dim3((unsigned)(NW / 8 / 256)), dim3(256), 0, stream>>>(Wi, wih, wil, (int)(NW / 8));
  split_bf16_kernel<<<dim3((unsigned)(NW / 8 / 256)), dim3(256), 0, stream>>>(Wh, whh, whl, (int)(NW / 8));

  // Phase 1: fused triple GEMM + gate math
  dim3 g1(HH / 64, (BB * SS) / 64);
  gates_wmma_kernel<<<g1, dim3(128), 0, stream>>>(xhi, xlo, wfh, wfl, wih, wil, whh, whl,
                                                  bf, bi, bh, lf, lv);

  // Phase 2: log-space scan
  dim3 g2((BB * HH) / SCAN_WAVES);
  scan_kernel<<<g2, dim3(64), 0, stream>>>(lf, lv, preh, out);
}
